// MPNN_85341000171639
// MI455X (gfx1250) — matrix operand order repaired and checked by reading, not compiled, not run on hardware
//
#include <hip/hip_runtime.h>
#include <math.h>

// ---------------------------------------------------------------------------
// MPNN forward for MI455X (gfx1250, wave32, WMMA).
// - never materialize ew[E,64,64] (1.6GB -> 6.4GB HBM); recompute per 16-edge
//   tile with v_wmma_f32_16x16x32_bf16 from ew2[E,64] bf16 + We3^T (512KB).
// - all weight matrices pre-packed into WMMA B-fragment order so each lane's
//   fragment is one contiguous 32B load (global_load_b128 / ds_load_b128).
// - msg kernel double-buffers the 8KB We3^T i-slab in LDS using
//   global_load_async_to_lds_b128 (ASYNCcnt) when available, overlapping the
//   weight staging with the WMMA stream; sync fallback otherwise.
// ---------------------------------------------------------------------------

namespace {
constexpr int N_NODES = 25000;
constexpr int N_EDGES = 100000;
constexpr int H       = 64;
constexpr int NODE_IN = 32;
constexpr int EDGE_IN = 16;
constexpr int NLAYERS = 3;
}

typedef __attribute__((ext_vector_type(16))) __bf16 v16bf;
typedef __attribute__((ext_vector_type(8)))  __bf16 v8bf;
typedef __attribute__((ext_vector_type(8)))  float  v8f;
typedef __attribute__((ext_vector_type(4)))  float  v4f;
typedef __attribute__((ext_vector_type(4)))  int    v4i;

#if defined(__has_builtin)
#if __has_builtin(__builtin_amdgcn_global_load_async_to_lds_b128)
#define HAVE_ASYNC_LDS 1
#endif
#endif
#ifndef HAVE_ASYNC_LDS
#define HAVE_ASYNC_LDS 0
#endif

static __device__ __forceinline__ v8f zero8() {
  v8f z = {0.f,0.f,0.f,0.f,0.f,0.f,0.f,0.f};
  return z;
}

static __device__ __forceinline__ v8f wmma_bf16(v16bf a, v16bf b, v8f c) {
  // D = A(16x32 bf16) * B(32x16 bf16) + C(16x16 f32)
  return __builtin_amdgcn_wmma_f32_16x16x32_bf16(false, a, false, b,
                                                 (short)0, c, false, false);
}

static __device__ __forceinline__ void wave_lds_fence() {
  asm volatile("" ::: "memory");
  __builtin_amdgcn_wave_barrier();
  asm volatile("" ::: "memory");
}

// Stage one 32B chunk global->LDS. Async path uses the CDNA5 async-load
// engine (tracked by ASYNCcnt); fallback is a plain LDS store of a load.
static __device__ __forceinline__ void stage32(const v16bf* g, v16bf* l) {
#if HAVE_ASYNC_LDS
  // prototype: (int4 addrspace(1)*, int4 addrspace(3)*, imm offset, imm cpol)
  __attribute__((address_space(1))) v4i* gp =
      (__attribute__((address_space(1))) v4i*)g;
  __attribute__((address_space(3))) v4i* lp =
      (__attribute__((address_space(3))) v4i*)l;
  __builtin_amdgcn_global_load_async_to_lds_b128(gp, lp, 0, 0);
  __builtin_amdgcn_global_load_async_to_lds_b128(gp, lp, 16, 0);
#else
  *l = *g;
#endif
}
static __device__ __forceinline__ void stage_wait() {
#if HAVE_ASYNC_LDS
  asm volatile("s_wait_asynccnt 0" ::: "memory");
#endif
}

// ISA 7.12.2 lane layouts (wave32):
// A frag 16x32: lane = half*16 + m; elem j -> k = j + 8*half + (j>=8 ? 8 : 0)
//   => j=0..7 is contiguous K [8h, 8h+8), j=8..15 contiguous K [16+8h, 24+8h)
// B frag 32x16: lane = half*16 + n; elem j -> k = j + 8*half + (j>=8 ? 8 : 0) (same as A)
// C frag 16x16: VGPR r -> row = r + 8*half, col = lane&15

// --- A fragment from row-major bf16 global (2 x 16B vector loads) ----------
static __device__ __forceinline__ v16bf a_frag_bf16g(const __bf16* base, int ld,
                                                     long row0, long rowmax,
                                                     int kbase) {
  int lane = threadIdx.x & 31, half = lane >> 4, m = lane & 15;
  long row = row0 + m; if (row > rowmax) row = rowmax;
  const __bf16* p = base + row * (long)ld + kbase + (half << 3);
  v8bf lo = *(const v8bf*)p;
  v8bf hi = *(const v8bf*)(p + 16);
  return __builtin_shufflevector(lo, hi, 0,1,2,3,4,5,6,7,
                                         8,9,10,11,12,13,14,15);
}

// --- A fragment from row-major f32 global, K zero-padded to 32 -------------
static __device__ __forceinline__ v16bf a_frag_f32g(const float* base, int ld,
                                                    long row0, long rowmax,
                                                    int kmax) {
  int lane = threadIdx.x & 31, half = lane >> 4, m = lane & 15;
  long row = row0 + m; if (row > rowmax) row = rowmax;
  const float* p = base + row * (long)ld + (half << 3);
  v4f a0 = *(const v4f*)p, a1 = *(const v4f*)(p + 4);
  v4f a2 = {0.f,0.f,0.f,0.f}, a3 = {0.f,0.f,0.f,0.f};
  if (16 + (half << 3) < kmax) {            // upper K chunk exists
    a2 = *(const v4f*)(p + 16);
    a3 = *(const v4f*)(p + 20);
  }
  v16bf a;
#pragma unroll
  for (int j = 0; j < 4; ++j) {
    a[j]      = (__bf16)a0[j];
    a[4 + j]  = (__bf16)a1[j];
    a[8 + j]  = (__bf16)a2[j];
    a[12 + j] = (__bf16)a3[j];
  }
  return a;
}

// --- A fragment from f32 LDS tile (4 x 16B ds loads + cvt) -----------------
static __device__ __forceinline__ v16bf a_frag_ldsf(const float* t, int ld,
                                                    int kbase) {
  int lane = threadIdx.x & 31, half = lane >> 4, m = lane & 15;
  const float* p = t + m * ld + kbase + (half << 3);
  v4f a0 = *(const v4f*)p,        a1 = *(const v4f*)(p + 4);
  v4f a2 = *(const v4f*)(p + 16), a3 = *(const v4f*)(p + 20);
  v16bf a;
#pragma unroll
  for (int j = 0; j < 4; ++j) {
    a[j]      = (__bf16)a0[j];
    a[4 + j]  = (__bf16)a1[j];
    a[8 + j]  = (__bf16)a2[j];
    a[12 + j] = (__bf16)a3[j];
  }
  return a;
}

// --- B fragment from pre-packed weights: one 32B load per lane -------------
// Packed layout: tile (kt, nt) -> 32 lanes x 16 bf16 contiguous.
static __device__ __forceinline__ v16bf b_frag_pk(const __bf16* Wp, int ntiles,
                                                  int kt, int nt) {
  int lane = threadIdx.x & 31;
  return *(const v16bf*)(Wp + (((long)(kt * ntiles + nt) * 32 + lane) << 4));
}

// ---------------------------------------------------------------------------
// Weight packing: dst fragment-ordered bf16.
//  B[k][n] = transpose ? src[n*fan_in + k] : src[k*Ncols + n]; 0 for k>=fan_in
// ---------------------------------------------------------------------------
__global__ void prep_pack(__bf16* dst, const float* src, int Ncols, int krows,
                          int fan_in, int transpose) {
  int idx = blockIdx.x * 256 + threadIdx.x;
  if (idx >= krows * Ncols) return;
  int j = idx & 15;
  int lane = (idx >> 4) & 31;
  int t = idx >> 9;
  int ntiles = Ncols >> 4;
  int nt = t % ntiles, kt = t / ntiles;
  int k = kt * 32 + ((lane >> 4) << 3) + (j & 7) + ((j >> 3) << 4);
  int n = nt * 16 + (lane & 15);
  float v = 0.f;
  if (k < fan_in)
    v = transpose ? src[(long)n * fan_in + k] : src[(long)k * Ncols + n];
  dst[idx] = (__bf16)v;
}

// We3^T packed per i-slab: We3p[i][kt(2)][nt(4)][lane(32)][j(16)]
//   k = kt*32 + (lane>>4)*8 + (j&7) + (j>>3)*16, col = i*64 + nt*16 + (lane&15)
//   value = We3[col][k]   (We3: [4096][64] row-major)
__global__ void prep_pack_we3(__bf16* dst, const float* src) {
  int idx = blockIdx.x * 256 + threadIdx.x;
  if (idx >= 64 * 4096) return;
  int j = idx & 15, lane = (idx >> 4) & 31;
  int t = idx >> 9;                       // i*8 + kt*4 + nt
  int nt = t & 3, kt = (t >> 2) & 1, i = t >> 3;
  int k = kt * 32 + ((lane >> 4) << 3) + (j & 7) + ((j >> 3) << 4);
  int col = i * 64 + nt * 16 + (lane & 15);
  dst[idx] = (__bf16)src[(long)col * 64 + k];
}

// ---------------------------------------------------------------------------
// Node MLP: h = (relu(relu(x@Wn1^T+b1)@Wn2^T+b2))@Wn3^T+b3.  1 wave = 16 rows.
// ---------------------------------------------------------------------------
__global__ void __launch_bounds__(256)
node_mlp_kernel(const float* __restrict__ x,
                const __bf16* __restrict__ Wn1p, const float* __restrict__ bn1,
                const __bf16* __restrict__ Wn2p, const float* __restrict__ bn2,
                const __bf16* __restrict__ Wn3p, const float* __restrict__ bn3,
                float* __restrict__ hF, __bf16* __restrict__ hB) {
  __shared__ __attribute__((aligned(16))) float T[8][16][68];
  int tid = threadIdx.x, w = tid >> 5, lane = tid & 31;
  int half = lane >> 4, n = lane & 15;
  long tile0 = ((long)blockIdx.x * 8 + w) * 16;
  float* t = &T[w][0][0];

  // layer 1: K = 32 (NODE_IN)
  v16bf a = a_frag_f32g(x, NODE_IN, tile0, N_NODES - 1, NODE_IN);
#pragma unroll
  for (int nb = 0; nb < 4; ++nb) {
    v8f c = wmma_bf16(a, b_frag_pk(Wn1p, 4, 0, nb), zero8());
#pragma unroll
    for (int r = 0; r < 8; ++r) {
      int row = r + half * 8, col = nb * 16 + n;
      t[row * 68 + col] = fmaxf(c[r] + bn1[col], 0.f);
    }
  }
  wave_lds_fence();

  // layer 2: K = 64
  v16bf a0 = a_frag_ldsf(t, 68, 0), a1 = a_frag_ldsf(t, 68, 32);
  v8f c2[4];
#pragma unroll
  for (int nb = 0; nb < 4; ++nb) {
    v8f c = wmma_bf16(a0, b_frag_pk(Wn2p, 4, 0, nb), zero8());
    c2[nb] = wmma_bf16(a1, b_frag_pk(Wn2p, 4, 1, nb), c);
  }
  wave_lds_fence();
#pragma unroll
  for (int nb = 0; nb < 4; ++nb)
#pragma unroll
    for (int r = 0; r < 8; ++r) {
      int row = r + half * 8, col = nb * 16 + n;
      t[row * 68 + col] = fmaxf(c2[nb][r] + bn2[col], 0.f);
    }
  wave_lds_fence();

  // layer 3 (no relu) -> global (f32 + bf16)
  a0 = a_frag_ldsf(t, 68, 0); a1 = a_frag_ldsf(t, 68, 32);
#pragma unroll
  for (int nb = 0; nb < 4; ++nb) {
    v8f c = wmma_bf16(a0, b_frag_pk(Wn3p, 4, 0, nb), zero8());
    c = wmma_bf16(a1, b_frag_pk(Wn3p, 4, 1, nb), c);
#pragma unroll
    for (int r = 0; r < 8; ++r) {
      int row = r + half * 8, col = nb * 16 + n;
      long grow = tile0 + row;
      if (grow < N_NODES) {
        float v = c[r] + bn3[col];
        hF[grow * H + col] = v;
        hB[grow * H + col] = (__bf16)v;
      }
    }
  }
}

// ---------------------------------------------------------------------------
// Edge MLP stages 1+2: ew2 = relu(relu(ea@We1^T+b1)@We2^T+b2)  (bf16 [E,64])
// ---------------------------------------------------------------------------
__global__ void __launch_bounds__(256)
edge_mlp_kernel(const float* __restrict__ ea,
                const __bf16* __restrict__ We1p, const float* __restrict__ be1,
                const __bf16* __restrict__ We2p, const float* __restrict__ be2,
                __bf16* __restrict__ ew2) {
  __shared__ __attribute__((aligned(16))) float T[8][16][68];
  int tid = threadIdx.x, w = tid >> 5, lane = tid & 31;
  int half = lane >> 4, n = lane & 15;
  long tile0 = ((long)blockIdx.x * 8 + w) * 16;
  float* t = &T[w][0][0];

  // layer 1: K = 16, zero-padded to 32 (We1p K rows 16..31 are zero)
  v16bf a = a_frag_f32g(ea, EDGE_IN, tile0, N_EDGES - 1, EDGE_IN);
#pragma unroll
  for (int nb = 0; nb < 4; ++nb) {
    v8f c = wmma_bf16(a, b_frag_pk(We1p, 4, 0, nb), zero8());
#pragma unroll
    for (int r = 0; r < 8; ++r) {
      int row = r + half * 8, col = nb * 16 + n;
      t[row * 68 + col] = fmaxf(c[r] + be1[col], 0.f);
    }
  }
  wave_lds_fence();

  v16bf a0 = a_frag_ldsf(t, 68, 0), a1 = a_frag_ldsf(t, 68, 32);
#pragma unroll
  for (int nb = 0; nb < 4; ++nb) {
    v8f c = wmma_bf16(a0, b_frag_pk(We2p, 4, 0, nb), zero8());
    c = wmma_bf16(a1, b_frag_pk(We2p, 4, 1, nb), c);
#pragma unroll
    for (int r = 0; r < 8; ++r) {
      int row = r + half * 8, col = nb * 16 + n;
      long e = tile0 + row;
      if (e < N_EDGES) ew2[e * H + col] = (__bf16)fmaxf(c[r] + be2[col], 0.f);
    }
  }
}

// ---------------------------------------------------------------------------
// Message + scatter:
//   msg[e,o] = sum_i G[e,i]*( sum_k ew2[e,k]*We3^T[k,i*64+o] ) + (G@Be3)[e,o]
// 16-edge tile per wave; 8 waves share the 8KB We3^T i-slab, double-buffered
// in LDS via async global->LDS (ASYNCcnt) so staging overlaps the WMMAs.
// ---------------------------------------------------------------------------
__global__ void __launch_bounds__(256)
msg_kernel(const float* __restrict__ hF, const __bf16* __restrict__ ew2,
           const __bf16* __restrict__ We3p, const __bf16* __restrict__ Be3p,
           const int* __restrict__ edge_index, float* __restrict__ aggr) {
  __shared__ __attribute__((aligned(16))) float Gf[8][16][68];
  __shared__ v16bf W3s[2][256];       // 2 x 8KB slab, fragment order
  __shared__ int   srcS[8][16];
  __shared__ int   dstS[8][16];

  int tid = threadIdx.x, w = tid >> 5, lane = tid & 31;
  int half = lane >> 4, nn = lane & 15;
  long ebase = ((long)blockIdx.x * 8 + w) * 16;
  const v16bf* W3g = (const v16bf*)We3p;   // 256 v16bf per i-slab

  if (lane < 16) {
    long e = ebase + lane; if (e > N_EDGES - 1) e = N_EDGES - 1;
    srcS[w][lane] = edge_index[e];
    dstS[w][lane] = edge_index[N_EDGES + e];
  }
  // prologue: start staging slab 0 while we gather G and build fragments
  stage32(W3g + tid, &W3s[0][tid]);
  wave_lds_fence();
  for (int idx = lane; idx < 16 * 16; idx += 32) {     // 16 rows x 16 v4f
    int r = idx >> 4, c4 = idx & 15;
    *(v4f*)&Gf[w][r][c4 * 4] = *(const v4f*)&hF[(long)srcS[w][r] * H + c4 * 4];
  }
  wave_lds_fence();

  v16bf ae0 = a_frag_bf16g(ew2, H, ebase, N_EDGES - 1, 0);
  v16bf ae1 = a_frag_bf16g(ew2, H, ebase, N_EDGES - 1, 32);
  v16bf ag0 = a_frag_ldsf(&Gf[w][0][0], 68, 0);
  v16bf ag1 = a_frag_ldsf(&Gf[w][0][0], 68, 32);

  // bias path: msg += G @ Be3
  v8f macc[4];
#pragma unroll
  for (int nb = 0; nb < 4; ++nb) {
    v8f c = wmma_bf16(ag0, b_frag_pk(Be3p, 4, 0, nb), zero8());
    macc[nb] = wmma_bf16(ag1, b_frag_pk(Be3p, 4, 1, nb), c);
  }

  stage_wait();
  __syncthreads();                       // slab 0 visible to all waves

  for (int i = 0; i < H; ++i) {
    int buf = i & 1;
    if (i + 1 < H)                        // overlap next slab with compute
      stage32(W3g + (long)(i + 1) * 256 + tid, &W3s[buf ^ 1][tid]);

    float g[8];
#pragma unroll
    for (int r = 0; r < 8; ++r) g[r] = Gf[w][r + half * 8][i];
#pragma unroll
    for (int nb = 0; nb < 4; ++nb) {
      v8f c = wmma_bf16(ae0, W3s[buf][nb * 32 + lane], zero8());
      c = wmma_bf16(ae1, W3s[buf][(4 + nb) * 32 + lane], c);
#pragma unroll
      for (int r = 0; r < 8; ++r) macc[nb][r] = fmaf(g[r], c[r], macc[nb][r]);
    }

    stage_wait();                         // retire this wave's async stores
    __syncthreads();                      // next slab visible; buf reusable
  }

  // scatter-add msg -> aggr[dst]  (global_atomic_add_f32)
#pragma unroll
  for (int nb = 0; nb < 4; ++nb)
#pragma unroll
    for (int r = 0; r < 8; ++r) {
      int row = r + half * 8;
      long e = ebase + row;
      if (e < N_EDGES) {
        int d = dstS[w][row];
        unsafeAtomicAdd(&aggr[(long)d * H + nb * 16 + nn], macc[nb][r]);
      }
    }
}

// ---------------------------------------------------------------------------
// Conv-combine + GRU cell:  m = aggr + h@root + cb;  hidden = GRU(m, hidden)
// ---------------------------------------------------------------------------
__global__ void __launch_bounds__(128)
gru_kernel(const float* __restrict__ aggr, const __bf16* __restrict__ hBin,
           const __bf16* __restrict__ rootsP, const float* __restrict__ conv_bias,
           int l, const __bf16* __restrict__ Wihp, const __bf16* __restrict__ Whhp,
           const float* __restrict__ bih, const float* __restrict__ bhh,
           float* __restrict__ hid, float* __restrict__ hF,
           __bf16* __restrict__ hBout) {
  __shared__ __attribute__((aligned(16))) float HID[4][16][68];
  __shared__ __attribute__((aligned(16))) float MM[4][16][68];
  int tid = threadIdx.x, w = tid >> 5, lane = tid & 31;
  int half = lane >> 4, n = lane & 15;
  long tile0 = ((long)blockIdx.x * 4 + w) * 16;
  const __bf16* Rl = rootsP + (long)l * H * H;
  const float* cb = conv_bias + (long)l * H;

  for (int idx = lane; idx < 16 * 16; idx += 32) {
    int r = idx >> 4, c4 = idx & 15;
    long row = tile0 + r; if (row > N_NODES - 1) row = N_NODES - 1;
    *(v4f*)&HID[w][r][c4 * 4] = *(const v4f*)&hid[row * H + c4 * 4];
  }
  v16bf ah0 = a_frag_bf16g(hBin, H, tile0, N_NODES - 1, 0);
  v16bf ah1 = a_frag_bf16g(hBin, H, tile0, N_NODES - 1, 32);
  wave_lds_fence();

  // m = aggr + h @ roots[l] + cb
#pragma unroll
  for (int nb = 0; nb < 4; ++nb) {
    v8f c = wmma_bf16(ah0, b_frag_pk(Rl, 4, 0, nb), zero8());
    c = wmma_bf16(ah1, b_frag_pk(Rl, 4, 1, nb), c);
#pragma unroll
    for (int r = 0; r < 8; ++r) {
      int row = r + half * 8, col = nb * 16 + n;
      long grow = tile0 + row; if (grow > N_NODES - 1) grow = N_NODES - 1;
      MM[w][row][col] = c[r] + aggr[grow * H + col] + cb[col];
    }
  }
  wave_lds_fence();

  v16bf am0 = a_frag_ldsf(&MM[w][0][0], 68, 0);
  v16bf am1 = a_frag_ldsf(&MM[w][0][0], 68, 32);
  v16bf ad0 = a_frag_ldsf(&HID[w][0][0], 68, 0);
  v16bf ad1 = a_frag_ldsf(&HID[w][0][0], 68, 32);

#pragma unroll
  for (int nb = 0; nb < 4; ++nb) {
    int col0 = nb * 16;
    v8f ir = wmma_bf16(am0, b_frag_pk(Wihp, 12, 0, nb), zero8());
    ir = wmma_bf16(am1, b_frag_pk(Wihp, 12, 1, nb), ir);
    v8f iz = wmma_bf16(am0, b_frag_pk(Wihp, 12, 0, 4 + nb), zero8());
    iz = wmma_bf16(am1, b_frag_pk(Wihp, 12, 1, 4 + nb), iz);
    v8f in_ = wmma_bf16(am0, b_frag_pk(Wihp, 12, 0, 8 + nb), zero8());
    in_ = wmma_bf16(am1, b_frag_pk(Wihp, 12, 1, 8 + nb), in_);
    v8f hr = wmma_bf16(ad0, b_frag_pk(Whhp, 12, 0, nb), zero8());
    hr = wmma_bf16(ad1, b_frag_pk(Whhp, 12, 1, nb), hr);
    v8f hz = wmma_bf16(ad0, b_frag_pk(Whhp, 12, 0, 4 + nb), zero8());
    hz = wmma_bf16(ad1, b_frag_pk(Whhp, 12, 1, 4 + nb), hz);
    v8f hn = wmma_bf16(ad0, b_frag_pk(Whhp, 12, 0, 8 + nb), zero8());
    hn = wmma_bf16(ad1, b_frag_pk(Whhp, 12, 1, 8 + nb), hn);
#pragma unroll
    for (int r = 0; r < 8; ++r) {
      int row = r + half * 8, col = col0 + n;
      long grow = tile0 + row;
      float rg = 1.f / (1.f + __expf(-(ir[r] + bih[col] + hr[r] + bhh[col])));
      float zg = 1.f / (1.f + __expf(-(iz[r] + bih[H + col] + hz[r] + bhh[H + col])));
      float ng = tanhf(in_[r] + bih[2 * H + col] + rg * (hn[r] + bhh[2 * H + col]));
      float hv = (1.f - zg) * ng + zg * HID[w][row][col];
      if (grow < N_NODES) {
        hid[grow * H + col] = hv;
        hF[grow * H + col] = hv;
        hBout[grow * H + col] = (__bf16)hv;
      }
    }
  }
}

// ---------------------------------------------------------------------------
extern "C" void kernel_launch(void* const* d_in, const int* in_sizes, int n_in,
                              void* d_out, int out_size, void* d_ws,
                              size_t ws_size, hipStream_t stream) {
  const float* x    = (const float*)d_in[0];
  const float* ea   = (const float*)d_in[1];
  const float* Wn1  = (const float*)d_in[2];  const float* bn1 = (const float*)d_in[3];
  const float* Wn2  = (const float*)d_in[4];  const float* bn2 = (const float*)d_in[5];
  const float* Wn3  = (const float*)d_in[6];  const float* bn3 = (const float*)d_in[7];
  const float* We1  = (const float*)d_in[8];  const float* be1 = (const float*)d_in[9];
  const float* We2  = (const float*)d_in[10]; const float* be2 = (const float*)d_in[11];
  const float* We3  = (const float*)d_in[12]; const float* be3 = (const float*)d_in[13];
  const float* roots = (const float*)d_in[14];
  const float* conv_bias = (const float*)d_in[15];
  const float* Wih  = (const float*)d_in[16]; const float* Whh = (const float*)d_in[17];
  const float* bih  = (const float*)d_in[18]; const float* bhh = (const float*)d_in[19];
  const int* edge_index = (const int*)d_in[20];

  char* wsb = (char*)d_ws;
  size_t off = 0;
  auto take = [&](size_t bytes) -> char* {
    char* p = wsb + off;
    off = (off + bytes + 255) & ~(size_t)255;
    return p;
  };
  float*  hF    = (float*)take((size_t)N_NODES * H * 4);
  float*  hid   = (float*)take((size_t)N_NODES * H * 4);
  float*  aggr  = (float*)take((size_t)N_NODES * H * 4);
  __bf16* hB    = (__bf16*)take((size_t)N_NODES * H * 2);
  __bf16* ew2   = (__bf16*)take((size_t)N_EDGES * H * 2);
  __bf16* Wn1p  = (__bf16*)take(32 * 64 * 2);
  __bf16* Wn2p  = (__bf16*)take(64 * 64 * 2);
  __bf16* Wn3p  = (__bf16*)take(64 * 64 * 2);
  __bf16* We1p  = (__bf16*)take(32 * 64 * 2);
  __bf16* We2p  = (__bf16*)take(64 * 64 * 2);
  __bf16* We3p  = (__bf16*)take((size_t)64 * 4096 * 2);
  __bf16* Be3p  = (__bf16*)take(4096 * 2);
  __bf16* rootsP = (__bf16*)take(3 * 4096 * 2);
  __bf16* Wihp  = (__bf16*)take(64 * 192 * 2);
  __bf16* Whhp  = (__bf16*)take(64 * 192 * 2);

  // --- weight packing (bf16, WMMA B-fragment order) ---
  prep_pack<<<(32 * 64 + 255) / 256, 256, 0, stream>>>(Wn1p, Wn1, 64, 32, 32, 1);
  prep_pack<<<(64 * 64 + 255) / 256, 256, 0, stream>>>(Wn2p, Wn2, 64, 64, 64, 1);
  prep_pack<<<(64 * 64 + 255) / 256, 256, 0, stream>>>(Wn3p, Wn3, 64, 64, 64, 1);
  prep_pack<<<(32 * 64 + 255) / 256, 256, 0, stream>>>(We1p, We1, 64, 32, 16, 1);
  prep_pack<<<(64 * 64 + 255) / 256, 256, 0, stream>>>(We2p, We2, 64, 64, 64, 1);
  prep_pack<<<(64 * 192 + 255) / 256, 256, 0, stream>>>(Wihp, Wih, 192, 64, 64, 1);
  prep_pack<<<(64 * 192 + 255) / 256, 256, 0, stream>>>(Whhp, Whh, 192, 64, 64, 1);
  prep_pack<<<(64 * 64 + 255) / 256, 256, 0, stream>>>(Be3p, be3, 64, 64, 64, 0);
  for (int l = 0; l < NLAYERS; ++l)
    prep_pack<<<(64 * 64 + 255) / 256, 256, 0, stream>>>(
        rootsP + (size_t)l * 4096, roots + (size_t)l * 4096, 64, 64, 64, 0);
  prep_pack_we3<<<(64 * 4096 + 255) / 256, 256, 0, stream>>>(We3p, We3);

  constexpr int node_tiles = (N_NODES + 15) / 16;     // 1563
  constexpr int edge_tiles = (N_EDGES + 15) / 16;     // 6250
  constexpr int node_blocks8 = (node_tiles + 7) / 8;  // 196
  constexpr int edge_blocks8 = (edge_tiles + 7) / 8;  // 782
  constexpr int node_blocks4 = (node_tiles + 3) / 4;  // 391

  node_mlp_kernel<<<node_blocks8, 256, 0, stream>>>(x, Wn1p, bn1, Wn2p, bn2,
                                                    Wn3p, bn3, hF, hB);
  edge_mlp_kernel<<<edge_blocks8, 256, 0, stream>>>(ea, We1p, be1, We2p, be2,
                                                    ew2);
  (void)hipMemsetAsync(hid, 0, (size_t)N_NODES * H * sizeof(float), stream);

  for (int l = 0; l < NLAYERS; ++l) {
    (void)hipMemsetAsync(aggr, 0, (size_t)N_NODES * H * sizeof(float), stream);
    msg_kernel<<<edge_blocks8, 256, 0, stream>>>(hF, ew2, We3p, Be3p,
                                                 edge_index, aggr);
    gru_kernel<<<node_blocks4, 128, 0, stream>>>(aggr, hB, rootsP, conv_bias,
                                                 l, Wihp, Whhp, bih, bhh,
                                                 hid, hF, hB);
  }

  (void)hipMemcpyAsync(d_out, hF, (size_t)N_NODES * H * sizeof(float),
                       hipMemcpyDeviceToDevice, stream);
}
